// NOzDEVJVmkXCBbrgeucbWnojtgGvnSHf_84980222919369
// MI455X (gfx1250) — matrix operand order repaired and checked by reading, not compiled, not run on hardware
//
#include <hip/hip_runtime.h>
#include <hip/hip_bf16.h>

#define CC   512
#define HWN  9216
#define NG   32
#define CPG  16
#define JT   256
#define EPSV 1e-6f

#define AS1 __attribute__((address_space(1)))
#define AS3 __attribute__((address_space(3)))

#if defined(__has_builtin)
#if __has_builtin(__builtin_amdgcn_global_load_async_to_lds_b128)
#define ASYNC_LDS 1
#endif
#endif

typedef _Float16 h16v __attribute__((ext_vector_type(16)));
typedef _Float16 h8v  __attribute__((ext_vector_type(8)));
typedef float    f8v  __attribute__((ext_vector_type(8)));
typedef int      i32x4 __attribute__((vector_size(16)));

// ---- WMMA fragment helpers (CDNA5 wave32 layouts, 16x16x32 f16) ----

// A (16x32, f16): lane m (m = lane&15) holds row m; first 8 halfs are
// K = hi*8..hi*8+7, second 8 halfs are K = 16+hi*8.. (hi = lane>>4).
__device__ __forceinline__ h16v ld_a(const _Float16* __restrict__ base, int ld, int lane) {
  int r = lane & 15, hi = lane >> 4;
  const _Float16* p = base + (size_t)r * ld + hi * 8;
  union { h16v v; h8v h[2]; } u;
  u.h[0] = *(const h8v*)(p);
  u.h[1] = *(const h8v*)(p + 16);
  return u.v;
}

// B (32x16, f16): lane holds column n = lane&15; first 8 halfs K = hi*8.., second 8 K = 16+hi*8..
// `base` points into column-major storage B[n][k] (row n contiguous over k).
__device__ __forceinline__ h16v ld_b(const _Float16* __restrict__ base, int ld, int lane) {
  int n = lane & 15, hi = lane >> 4;
  const _Float16* p = base + (size_t)n * ld + hi * 8;
  union { h16v v; h8v h[2]; } u;
  u.h[0] = *(const h8v*)(p);
  u.h[1] = *(const h8v*)(p + 16);
  return u.v;
}

__device__ __forceinline__ f8v wmma_f16(h16v a, h16v b, f8v c) {
  return __builtin_amdgcn_wmma_f32_16x16x32_f16(false, a, false, b, (short)0, c, false, false);
}

// ---- weight fp32 -> fp16 conversion (4 x [C,C]) ----
__global__ __launch_bounds__(256) void cvt_w_kernel(const float* __restrict__ w0,
                                                    const float* __restrict__ w1,
                                                    const float* __restrict__ w2,
                                                    const float* __restrict__ w3,
                                                    _Float16* __restrict__ dst) {
  int idx = blockIdx.x * 256 + threadIdx.x;           // 4*512*512 total
  int which = idx >> 18, off = idx & 0x3FFFF;         // C*C == 2^18
  const float* src = (which == 0) ? w0 : (which == 1) ? w1 : (which == 2) ? w2 : w3;
  dst[idx] = (_Float16)src[off];
}

// ---- GroupNorm stats: one block per group ----
__global__ __launch_bounds__(256) void gn_stats_kernel(const float* __restrict__ x,
                                                       float* __restrict__ stats) {
  __shared__ float ssum[256], ssq[256];
  int g = blockIdx.x, tid = threadIdx.x;
  const float* p = x + (size_t)g * CPG * HWN;
  const int n = CPG * HWN;
  float s = 0.f, q = 0.f;
  for (int i = tid; i < n; i += 256) { float v = p[i]; s += v; q += v * v; }
  ssum[tid] = s; ssq[tid] = q;
  __syncthreads();
  for (int st = 128; st > 0; st >>= 1) {
    if (tid < st) { ssum[tid] += ssum[tid + st]; ssq[tid] += ssq[tid + st]; }
    __syncthreads();
  }
  if (tid == 0) {
    float mean = ssum[0] / n;
    float var  = ssq[0] / n - mean * mean;
    stats[g]      = mean;
    stats[NG + g] = rsqrtf(var + EPSV);
  }
}

// ---- GroupNorm apply + transpose to hn[HW][C] f16 ----
__global__ __launch_bounds__(256) void gn_apply_kernel(const float* __restrict__ x,
                                                       const float* __restrict__ gamma,
                                                       const float* __restrict__ beta,
                                                       const float* __restrict__ stats,
                                                       _Float16* __restrict__ hn) {
  __shared__ float tile[32][33];
  int tx = threadIdx.x, ty = threadIdx.y;
  int i0 = blockIdx.x * 32, c0 = blockIdx.y * 32;
  for (int cc = ty; cc < 32; cc += 8)
    tile[cc][tx] = x[(size_t)(c0 + cc) * HWN + i0 + tx];   // coalesced over pixels
  __syncthreads();
  int c = c0 + tx, g = c >> 4;
  float mean = stats[g], rstd = stats[NG + g];
  float ga = gamma[c], be = beta[c];
  for (int ii = ty; ii < 32; ii += 8) {
    float v = (tile[tx][ii] - mean) * rstd * ga + be;
    hn[(size_t)(i0 + ii) * CC + c] = (_Float16)v;          // coalesced over channels
  }
}

// ---- generic WMMA GEMM: D(MxN) = A(MxK) * B^T + bias, K = 512 ----
// Block = 128 threads (4 waves). Block tile 128x64; wave tile 32x64.
// B panel (64 x 32, f16) is staged in LDS (async copy, double buffered) and
// shared by all 4 waves.
// A row-major [m][k] (lda), B stored as [n][k] (ldb).
// mode 0/1: f16 store row-major [m][n] (ldd); bias indexed by col (0) or row (1)
// mode 2  : final f32 store transposed out[n*HW + m] = acc + bias[n] + xres[n*HW + m]
__global__ __launch_bounds__(128) void gemm_kernel(
    const _Float16* __restrict__ A, int lda,
    const _Float16* __restrict__ B, int ldb,
    const float* __restrict__ bias, int bias_on_rows, float alpha,
    _Float16* __restrict__ D, int ldd,
    float* __restrict__ Dfin, const float* __restrict__ xres, int mode) {
  __shared__ _Float16 Bsh[2][64 * 32];
  int tid = threadIdx.x, lane = tid & 31, wave = tid >> 5;
  int m0 = blockIdx.x * 128 + wave * 32;
  int n0 = blockIdx.y * 64;

  // cooperative stage of B panel [n0..n0+64) x [kk..kk+32) into Bsh[bb]
  auto stage = [&](int kk, int bb) {
#pragma unroll
    for (int it = 0; it < 2; ++it) {
      int ch = tid + it * 128;             // 256 x 16-byte chunks
      int row = ch >> 2, part = ch & 3;
      const _Float16* src = B + (size_t)(n0 + row) * ldb + kk + part * 8;
      _Float16* dst = &Bsh[bb][row * 32 + part * 8];
#if defined(ASYNC_LDS)
      __builtin_amdgcn_global_load_async_to_lds_b128((AS1 i32x4*)src, (AS3 i32x4*)dst, 0, 0);
#else
      *(h8v*)dst = *(const h8v*)src;
#endif
    }
  };

  f8v acc[2][4] = {};
  const _Float16* abase0 = A + (size_t)m0 * lda;
  const _Float16* abase1 = A + (size_t)(m0 + 16) * lda;

  stage(0, 0);
#if defined(ASYNC_LDS)
  asm volatile("s_wait_asynccnt 0x0" ::: "memory");
#endif
  __syncthreads();

  int buf = 0;
  for (int k0 = 0; k0 < CC; k0 += 32) {
    if (k0 + 32 < CC) stage(k0 + 32, buf ^ 1);
    __builtin_prefetch(abase0 + k0 + 64, 0, 0);
    h16v a0 = ld_a(abase0 + k0, lda, lane);
    h16v a1 = ld_a(abase1 + k0, lda, lane);
#pragma unroll
    for (int s = 0; s < 4; ++s) {
      h16v b = ld_b(&Bsh[buf][s * 16 * 32], 32, lane);   // from LDS, stride 32
      acc[0][s] = wmma_f16(a0, b, acc[0][s]);
      acc[1][s] = wmma_f16(a1, b, acc[1][s]);
    }
#if defined(ASYNC_LDS)
    asm volatile("s_wait_asynccnt 0x0" ::: "memory");
#endif
    __syncthreads();
    buf ^= 1;
  }

  int ncol = lane & 15, mofs = (lane >> 4) * 8;
#pragma unroll
  for (int rb = 0; rb < 2; ++rb) {
#pragma unroll
    for (int s = 0; s < 4; ++s) {
      int col = n0 + s * 16 + ncol;
#pragma unroll
      for (int v = 0; v < 8; ++v) {
        int row = m0 + rb * 16 + v + mofs;
        float bval = bias_on_rows ? bias[row] : bias[col];
        float val = (acc[rb][s][v] + bval) * alpha;
        if (mode == 2)
          Dfin[(size_t)col * HWN + row] = val + xres[(size_t)col * HWN + row];
        else
          D[(size_t)row * ldd + col] = (_Float16)val;
      }
    }
  }
}

// ---- flash attention: block = 16 queries, 8 waves; wave owns 64 output channels ----
__global__ __launch_bounds__(256) void attn_kernel(const _Float16* __restrict__ Qs,
                                                   const _Float16* __restrict__ Kk,
                                                   const _Float16* __restrict__ Vc,
                                                   _Float16* __restrict__ Ot) {
  __shared__ float    S[16][JT];       // raw score chunk (16 x 256)
  __shared__ _Float16 P[16][JT];       // probabilities f16
  __shared__ float pmax[16][16], psum[16][16];
  __shared__ float mrow[16], lrow[16], rfac[16], mnew[16];

  int tid = threadIdx.x, lane = tid & 31, wave = tid >> 5;
  int q0 = blockIdx.x * 16;
  int ncol = lane & 15, mofs = (lane >> 4) * 8;

  if (tid < 16) { mrow[tid] = -3.0e38f; lrow[tid] = 0.f; }
  __syncthreads();

  f8v acc[4] = {};                      // O accum: 16 q-rows x 64 channels
  const _Float16* qbase = Qs + (size_t)q0 * CC;

  for (int jt = 0; jt < HWN; jt += JT) {
    // (1) scores for this wave's 32-key slice: S[0:16, wave*32 : wave*32+32)
    int jb = jt + wave * 32;
    f8v sc0 = {}, sc1 = {};
    for (int k0 = 0; k0 < CC; k0 += 32) {
      h16v a  = ld_a(qbase + k0, CC, lane);
      h16v b0 = ld_b(Kk + (size_t)jb * CC + k0, CC, lane);
      h16v b1 = ld_b(Kk + (size_t)(jb + 16) * CC + k0, CC, lane);
      sc0 = wmma_f16(a, b0, sc0);
      sc1 = wmma_f16(a, b1, sc1);
    }
#pragma unroll
    for (int v = 0; v < 8; ++v) {
      S[v + mofs][wave * 32 + ncol]      = sc0[v];
      S[v + mofs][wave * 32 + 16 + ncol] = sc1[v];
    }
    __syncthreads();

    // (2) per-row partial max (16 threads per row)
    {
      int row = tid >> 4, seg = tid & 15;
      float m = -3.0e38f;
#pragma unroll
      for (int k = 0; k < 16; ++k) m = fmaxf(m, S[row][seg * 16 + k]);
      pmax[row][seg] = m;
    }
    __syncthreads();
    if (tid < 16) {
      float mo = mrow[tid], mc = -3.0e38f;
      for (int k = 0; k < 16; ++k) mc = fmaxf(mc, pmax[tid][k]);
      float mn = fmaxf(mo, mc);
      mnew[tid] = mn; rfac[tid] = __expf(mo - mn); mrow[tid] = mn;
    }
    __syncthreads();

    // (3) P = exp(S - mnew), partial row sums
    {
      int row = tid >> 4, seg = tid & 15;
      float mn = mnew[row], s = 0.f;
#pragma unroll
      for (int k = 0; k < 16; ++k) {
        float e = __expf(S[row][seg * 16 + k] - mn);
        P[row][seg * 16 + k] = (_Float16)e;
        s += e;
      }
      psum[row][seg] = s;
    }
    __syncthreads();
    if (tid < 16) {
      float s = 0.f;
      for (int k = 0; k < 16; ++k) s += psum[tid][k];
      lrow[tid] = lrow[tid] * rfac[tid] + s;
    }
    __syncthreads();

    // (4) rescale O accum, then O += P(16xJT) * V(JT x 64) via WMMA
    float rloc[8];
#pragma unroll
    for (int v = 0; v < 8; ++v) rloc[v] = rfac[v + mofs];
#pragma unroll
    for (int s = 0; s < 4; ++s)
#pragma unroll
      for (int v = 0; v < 8; ++v) acc[s][v] *= rloc[v];

    for (int k0 = 0; k0 < JT; k0 += 32) {
      h16v a = ld_a(&P[0][0] + k0, JT, lane);   // from LDS
#pragma unroll
      for (int s = 0; s < 4; ++s) {
        int c0 = wave * 64 + s * 16;
        h16v b = ld_b(Vc + (size_t)c0 * HWN + jt + k0, HWN, lane);
        acc[s] = wmma_f16(a, b, acc[s]);
      }
    }
    __syncthreads();   // protect S/P/rfac before next chunk
  }

  // finalize: divide by row sums, store Ot[HW][C] f16
  float linv[8];
#pragma unroll
  for (int v = 0; v < 8; ++v) linv[v] = 1.0f / lrow[v + mofs];
#pragma unroll
  for (int s = 0; s < 4; ++s) {
    int c = wave * 64 + s * 16 + ncol;
#pragma unroll
    for (int v = 0; v < 8; ++v)
      Ot[(size_t)(q0 + v + mofs) * CC + c] = (_Float16)(acc[s][v] * linv[v]);
  }
}

extern "C" void kernel_launch(void* const* d_in, const int* in_sizes, int n_in,
                              void* d_out, int out_size, void* d_ws, size_t ws_size,
                              hipStream_t stream) {
  const float* x     = (const float*)d_in[0];
  const float* gamma = (const float*)d_in[1];
  const float* beta  = (const float*)d_in[2];
  const float* q_w   = (const float*)d_in[3];
  const float* q_b   = (const float*)d_in[4];
  const float* k_w   = (const float*)d_in[5];
  const float* k_b   = (const float*)d_in[6];
  const float* v_w   = (const float*)d_in[7];
  const float* v_b   = (const float*)d_in[8];
  const float* p_w   = (const float*)d_in[9];
  const float* p_b   = (const float*)d_in[10];
  float* out = (float*)d_out;

  char* ws = (char*)d_ws;
  const size_t mat = (size_t)HWN * CC * sizeof(_Float16);     // 9 MB per matrix
  _Float16* hn  = (_Float16*)(ws);
  _Float16* Qs  = (_Float16*)(ws + 1 * mat);  // [HW][C], pre-scaled
  _Float16* Kk  = (_Float16*)(ws + 2 * mat);  // [HW][C]
  _Float16* Vc  = (_Float16*)(ws + 3 * mat);  // [C][HW]
  _Float16* Ot  = (_Float16*)(ws + 4 * mat);  // [HW][C]
  _Float16* w16 = (_Float16*)(ws + 5 * mat);  // 4 x [C][C]
  float* stats  = (float*)(ws + 5 * mat + (size_t)4 * CC * CC * sizeof(_Float16));

  cvt_w_kernel<<<4096, 256, 0, stream>>>(q_w, k_w, v_w, p_w, w16);
  gn_stats_kernel<<<NG, 256, 0, stream>>>(x, stats);
  gn_apply_kernel<<<dim3(HWN / 32, CC / 32), dim3(32, 8), 0, stream>>>(x, gamma, beta, stats, hn);

  const float scale = 0.044194173824159216f;  // 512^-0.5 folded into Q

  // Q[i][o] = scale*(hn . q_w^T + b)
  gemm_kernel<<<dim3(HWN / 128, CC / 64), 128, 0, stream>>>(
      hn, CC, w16 + 0 * CC * CC, CC, q_b, 0, scale, Qs, CC, nullptr, nullptr, 0);
  // K[i][o]
  gemm_kernel<<<dim3(HWN / 128, CC / 64), 128, 0, stream>>>(
      hn, CC, w16 + 1 * CC * CC, CC, k_b, 0, 1.0f, Kk, CC, nullptr, nullptr, 0);
  // V[o][j] (channel-major): A = v_w rows, B columns = hn rows
  gemm_kernel<<<dim3(CC / 128, HWN / 64), 128, 0, stream>>>(
      w16 + 2 * CC * CC, CC, hn, CC, v_b, 1, 1.0f, Vc, HWN, nullptr, nullptr, 1);

  attn_kernel<<<HWN / 16, 256, 0, stream>>>(Qs, Kk, Vc, Ot);

  // proj + bias + residual -> out[C][HW] f32
  gemm_kernel<<<dim3(HWN / 128, CC / 64), 128, 0, stream>>>(
      Ot, CC, w16 + 3 * CC * CC, CC, p_b, 0, 1.0f, nullptr, 0, out, x, 2);
}
